// Qwen3Attention_22909355557316
// MI455X (gfx1250) — matrix operand order repaired and checked by reading, not compiled, not run on hardware
//
#include <hip/hip_runtime.h>

typedef __attribute__((ext_vector_type(16))) __bf16 v16bf;
typedef __attribute__((ext_vector_type(8)))  float  v8f;
typedef __attribute__((ext_vector_type(4)))  unsigned int su4;
typedef __attribute__((ext_vector_type(8)))  unsigned int su8;
typedef unsigned short u16;
typedef unsigned int   u32;
typedef unsigned long long u64;

union FragBF { v16bf v; uint4 q[2]; };

__device__ inline u16 f2bf(float f) {
    u32 u = __float_as_uint(f);
    return (u16)((u + 0x7FFFu + ((u >> 16) & 1u)) >> 16);   // round-to-nearest-even
}

// ---------------------------------------------------------------------------
// Tensor Data Mover helpers (CDNA5 TDM, CDNA5 ISA/8)
// Descriptor group0: [1:0]=count=1, [63:32]=lds_addr, [120:64]=global_addr,
//                    [127:126]=type=2.
// Descriptor group1: data_size=2B, pad_enable, pad_interval=16 DW,
//                    pad_amount=4 DW (=> 80 B LDS row stride), 2D tile.
// ---------------------------------------------------------------------------
__device__ inline su4 tdm_g0(u32 lds_addr, const void* gptr) {
    u64 ga = (u64)gptr;
    su4 r;
    r[0] = 1u;                                   // count=1, user mode
    r[1] = lds_addr;                             // LDS byte address
    r[2] = (u32)ga;                              // global_addr[31:0]
    r[3] = ((u32)(ga >> 32) & 0x01FFFFFFu) | (2u << 30);  // addr[56:32] | type=2
    return r;
}

__device__ inline su8 tdm_g1_tile32x128(int K) {
    su8 g;
    g[0] = (1u << 16)        // data_size = 1 -> 2 bytes
         | (1u << 20)        // pad_enable
         | (3u << 22)        // pad_interval code 3 = every 16 DWORDs (64 B row)
         | (3u << 25);       // pad_amount  code 3 = 4 DWORDs (16 B) -> 80 B stride
    g[1] = (32u & 0xFFFFu) << 16;        // tensor_dim0 = 32 (lo16 in [63:48])
    g[2] = (128u << 16);                 // tensor_dim0 hi=0 | tensor_dim1 = 128
    g[3] = (32u << 16);                  // tensor_dim1 hi=0 | tile_dim0 = 32
    g[4] = 128u;                         // tile_dim1 = 128 | tile_dim2 = 0
    g[5] = (u32)K;                       // tensor_dim0_stride = K elements
    g[6] = 0u;
    g[7] = 0u;
    return g;
}

__device__ inline void tdm_load(su4 g0, su8 g1) {
    asm volatile("tensor_load_to_lds %0, %1" :: "s"(g0), "s"(g1) : "memory");
}

// ---------------------------------------------------------------------------
// Elementwise fp32 -> bf16 convert (4 elements / thread)
// ---------------------------------------------------------------------------
__global__ __launch_bounds__(256) void k_f2bf(const float* __restrict__ in,
                                              u16* __restrict__ out) {
    int i = (blockIdx.x * 256 + threadIdx.x) * 4;
#pragma unroll
    for (int j = 0; j < 4; ++j) out[i + j] = f2bf(in[i + j]);
}

// Build concatenated [Wq; Wk; Wv] (4096 x 2048) in bf16
__global__ __launch_bounds__(256) void k_wcat(const float* __restrict__ q_w,
                                              const float* __restrict__ k_w,
                                              const float* __restrict__ v_w,
                                              u16* __restrict__ out) {
    int i   = (blockIdx.x * 256 + threadIdx.x) * 4;
    int row = i >> 11;
    int col = i & 2047;
    const float* src = (row < 2048) ? q_w + (size_t)row * 2048 + col
                     : (row < 3072) ? k_w + (size_t)(row - 2048) * 2048 + col
                                    : v_w + (size_t)(row - 3072) * 2048 + col;
#pragma unroll
    for (int j = 0; j < 4; ++j) out[i + j] = f2bf(src[j]);
}

// ---------------------------------------------------------------------------
// Tiled bf16 WMMA GEMM:  C[M,N] = A[M,K] * B[N,K]^T   (both row-major over K)
// Block tile 128x128, BK=32, 8 waves (4x2), wave tile 32x64.
// Tiles staged by the Tensor Data Mover (double-buffered): wave 0 issues
// tensor_load_to_lds for tile i+1 while all waves run WMMA on tile i.
// TDM hardware padding gives an 80 B LDS row stride (conflict-free b128 reads).
// ---------------------------------------------------------------------------
__global__ __launch_bounds__(256) void k_gemm(const u16* __restrict__ A,
                                              const u16* __restrict__ Bw,
                                              float* __restrict__ C,
                                              int M, int N, int K) {
    __shared__ u16 As[2][128 * 40];
    __shared__ u16 Bs[2][128 * 40];
    const int tid  = threadIdx.x, lane = tid & 31, wave = tid >> 5;
    const int wm   = wave >> 1, wn = wave & 1;
    const int mblk = blockIdx.y * 128, nblk = blockIdx.x * 128;

    v8f acc[2][4];
#pragma unroll
    for (int a = 0; a < 2; ++a)
#pragma unroll
        for (int n = 0; n < 4; ++n) acc[a][n] = {};

    const su8 g1 = tdm_g1_tile32x128(K);
    const u32 ldsA0 = (u32)(uintptr_t)&As[0][0], ldsA1 = (u32)(uintptr_t)&As[1][0];
    const u32 ldsB0 = (u32)(uintptr_t)&Bs[0][0], ldsB1 = (u32)(uintptr_t)&Bs[1][0];
    const u16* arow = A  + (size_t)mblk * K;
    const u16* brow = Bw + (size_t)nblk * K;

    if (wave == 0) {                       // prologue: DMA tile 0 into buffer 0
        tdm_load(tdm_g0(ldsA0, arow), g1);
        tdm_load(tdm_g0(ldsB0, brow), g1);
    }

    const int T = K >> 5;                  // K/32 k-tiles
    for (int it = 0; it < T; ++it) {
        const int p = it & 1;
        if (wave == 0) {
            if (it + 1 < T) {              // DMA next tile into the other buffer
                const int kb2 = (it + 1) * 32;
                tdm_load(tdm_g0(p ? ldsA0 : ldsA1, arow + kb2), g1);
                tdm_load(tdm_g0(p ? ldsB0 : ldsB1, brow + kb2), g1);
                __builtin_amdgcn_s_wait_tensorcnt(2);  // current tile complete
            } else {
                __builtin_amdgcn_s_wait_tensorcnt(0);
            }
        }
        __syncthreads();                   // publish tile `it` to all waves

        const u16* as = &As[p][0];
        const u16* bs = &Bs[p][0];
        FragBF af[2], bf[4];
#pragma unroll
        for (int a = 0; a < 2; ++a) {      // A-frag: lane=M, K split chunks
            int row = wm * 32 + a * 16 + (lane & 15);
            int off = ((lane >> 4) & 1) * 8;
            af[a].q[0] = *(const uint4*)&as[row * 40 + off];
            af[a].q[1] = *(const uint4*)&as[row * 40 + off + 16];
        }
#pragma unroll
        for (int n = 0; n < 4; ++n) {      // B-frag: lane=N, K split chunks
            int row = wn * 64 + n * 16 + (lane & 15);
            int off = ((lane >> 4) & 1) * 8;
            bf[n].q[0] = *(const uint4*)&bs[row * 40 + off];
            bf[n].q[1] = *(const uint4*)&bs[row * 40 + off + 16];
        }
#pragma unroll
        for (int a = 0; a < 2; ++a)
#pragma unroll
            for (int n = 0; n < 4; ++n)
                acc[a][n] = __builtin_amdgcn_wmma_f32_16x16x32_bf16(
                    false, af[a].v, false, bf[n].v, (short)0, acc[a][n], false, false);
        __syncthreads();                   // all reads done before overwrite
    }

    const int rowoff = ((lane >> 4) & 1) * 8, col = lane & 15;
#pragma unroll
    for (int a = 0; a < 2; ++a)
#pragma unroll
        for (int n = 0; n < 4; ++n)
#pragma unroll
            for (int r = 0; r < 8; ++r)
                C[(size_t)(mblk + wm * 32 + a * 16 + rowoff + r) * N +
                  (nblk + wn * 64 + n * 16 + col)] = acc[a][n][r];
}

// ---------------------------------------------------------------------------
// Per-(b,s) RMSNorm + RoPE; wave-per-head (wave32). Emits bf16:
//   Q [b,h,s,d], K [b,kv,s,d], V transposed [b,kv,d,s]
// ---------------------------------------------------------------------------
__global__ __launch_bounds__(256) void k_normrope(const float* __restrict__ qkv,
                                                  const float* __restrict__ pe,
                                                  const float* __restrict__ qnw,
                                                  const float* __restrict__ knw,
                                                  u16* __restrict__ qb,
                                                  u16* __restrict__ kb_,
                                                  u16* __restrict__ vtb) {
    const int bs   = blockIdx.x;
    const int b    = bs >> 11, s = bs & 2047;
    const int lane = threadIdx.x & 31, wave = threadIdx.x >> 5;
    const float* row = qkv + (size_t)bs * 4096;
    const float* per = pe  + (size_t)bs * 128;
    const float ca = per[lane],      cb = per[lane + 32];
    const float sa = per[lane + 64], sb = per[lane + 96];

    auto doHead = [&](const float* src, const float* nw, u16* dst) {
        float v0 = src[lane], v1 = src[lane + 32],
              v2 = src[lane + 64], v3 = src[lane + 96];
        float ss = v0 * v0 + v1 * v1 + v2 * v2 + v3 * v3;
#pragma unroll
        for (int m = 16; m >= 1; m >>= 1) ss += __shfl_xor(ss, m, 32);
        float rinv = rsqrtf(ss * (1.f / 128.f) + 1e-6f);
        float n0 = v0 * rinv * nw[lane],      n1 = v1 * rinv * nw[lane + 32];
        float n2 = v2 * rinv * nw[lane + 64], n3 = v3 * rinv * nw[lane + 96];
        dst[lane]      = f2bf(n0 * ca - n2 * sa);
        dst[lane + 32] = f2bf(n1 * cb - n3 * sb);
        dst[lane + 64] = f2bf(n0 * sa + n2 * ca);
        dst[lane + 96] = f2bf(n1 * sb + n3 * cb);
    };

#pragma unroll
    for (int j = 0; j < 2; ++j) {                       // 2 q-heads per wave
        int h = wave * 2 + j;
        doHead(row + h * 128, qnw, qb + ((size_t)(b * 16 + h) * 2048 + s) * 128);
    }
    doHead(row + 2048 + wave * 128, knw,               // 1 k-head per wave
           kb_ + ((size_t)(b * 8 + wave) * 2048 + s) * 128);

    const float* vsrc = row + 3072 + wave * 128;       // V -> transposed bf16
    u16* vdst = vtb + (size_t)(b * 8 + wave) * 128 * 2048;
#pragma unroll
    for (int j = 0; j < 4; ++j) {
        int d = lane + 32 * j;
        vdst[(size_t)d * 2048 + s] = f2bf(vsrc[d]);
    }
}

// ---------------------------------------------------------------------------
// Flash attention (causal, GQA rep=2). Block = (b, h, 128 q rows); 8 waves,
// each wave owns 16 q rows and streams 64-key blocks with online softmax.
// P goes through a per-wave padded LDS tile for C-layout -> A-layout.
// ---------------------------------------------------------------------------
__global__ __launch_bounds__(256) void k_attn(const u16* __restrict__ qb,
                                              const u16* __restrict__ kbuf,
                                              const u16* __restrict__ vtb,
                                              u16* __restrict__ attn) {
    __shared__ u16 Pl[8 * 16 * 72];      // per-wave 16x64 P tile, stride 72
    const int lane = threadIdx.x & 31, wave = threadIdx.x >> 5;
    const int qblk = blockIdx.x, h = blockIdx.y, b = blockIdx.z, kvh = h >> 1;

    const u16* qbase  = qb   + (size_t)(b * 16 + h)   * 2048 * 128;
    const u16* kbase  = kbuf + (size_t)(b * 8 + kvh)  * 2048 * 128;
    const u16* vtbase = vtb  + (size_t)(b * 8 + kvh)  * 128 * 2048;

    const int col = lane & 15, hi = (lane >> 4) & 1;
    const int qrow0 = qblk * 128 + wave * 16;

    FragBF qf[4];                         // Q rows kept resident: 16 x 128
#pragma unroll
    for (int c = 0; c < 4; ++c) {
        const u16* p = qbase + (size_t)(qrow0 + col) * 128 + c * 32 + hi * 8;
        qf[c].q[0] = *(const uint4*)p;
        qf[c].q[1] = *(const uint4*)(p + 16);
    }

    v8f oacc[8];
#pragma unroll
    for (int t = 0; t < 8; ++t) oacc[t] = {};
    float mst[8], lst[8];
#pragma unroll
    for (int r = 0; r < 8; ++r) { mst[r] = -__builtin_inff(); lst[r] = 0.f; }

    u16* Pb = Pl + wave * 16 * 72;
    const int rowg = qrow0 + hi * 8;                     // row of component r
    const int kend = ((qrow0 + 15) / 64 + 1) * 64;       // causal key bound
    const float scale = 0.08838834764831845f;            // 1/sqrt(128)

    for (int kb = 0; kb < kend; kb += 64) {
        v8f sacc[4];
#pragma unroll
        for (int n = 0; n < 4; ++n) sacc[n] = {};
#pragma unroll
        for (int n = 0; n < 4; ++n) {                    // S = Q K^T
            const u16* krow = kbase + (size_t)(kb + n * 16 + col) * 128 + hi * 8;
#pragma unroll
            for (int c = 0; c < 4; ++c) {
                FragBF kf;
                kf.q[0] = *(const uint4*)(krow + c * 32);
                kf.q[1] = *(const uint4*)(krow + c * 32 + 16);
                sacc[n] = __builtin_amdgcn_wmma_f32_16x16x32_bf16(
                    false, qf[c].v, false, kf.v, (short)0, sacc[n], false, false);
            }
        }
        float rm[8];                                     // row max (shuffle)
#pragma unroll
        for (int r = 0; r < 8; ++r) {
            float tm = -__builtin_inff();
#pragma unroll
            for (int n = 0; n < 4; ++n) {
                float sv = sacc[n][r] * scale;
                sacc[n][r] = sv;
                bool msk = (kb + n * 16 + col) > (rowg + r);
                tm = fmaxf(tm, msk ? -__builtin_inff() : sv);
            }
#pragma unroll
            for (int m = 8; m >= 1; m >>= 1) tm = fmaxf(tm, __shfl_xor(tm, m, 32));
            rm[r] = tm;
        }
#pragma unroll
        for (int r = 0; r < 8; ++r) {                    // online softmax
            float mo = mst[r];
            float mn = fmaxf(mo, rm[r]);
            float corr = __expf(mo - mn);
            float ps = 0.f;
#pragma unroll
            for (int n = 0; n < 4; ++n) {
                bool msk = (kb + n * 16 + col) > (rowg + r);
                float p = msk ? 0.f : __expf(sacc[n][r] - mn);
                ps += p;
                Pb[(r + hi * 8) * 72 + n * 16 + col] = f2bf(p);
            }
#pragma unroll
            for (int m = 8; m >= 1; m >>= 1) ps += __shfl_xor(ps, m, 32);
            lst[r] = lst[r] * corr + ps;
            mst[r] = mn;
#pragma unroll
            for (int t = 0; t < 8; ++t) oacc[t][r] *= corr;
        }
        // O += P V   (per-wave LDS ops are in-order: no barrier needed)
#pragma unroll
        for (int c2 = 0; c2 < 2; ++c2) {
            FragBF pf;
            const u16* pp = Pb + (lane & 15) * 72 + c2 * 32 + hi * 8;
            pf.q[0] = *(const uint4*)pp;
            pf.q[1] = *(const uint4*)(pp + 16);
#pragma unroll
            for (int t = 0; t < 8; ++t) {
                FragBF vf;
                const u16* vp = vtbase + (size_t)(t * 16 + col) * 2048 +
                                kb + c2 * 32 + hi * 8;
                vf.q[0] = *(const uint4*)vp;
                vf.q[1] = *(const uint4*)(vp + 16);
                oacc[t] = __builtin_amdgcn_wmma_f32_16x16x32_bf16(
                    false, pf.v, false, vf.v, (short)0, oacc[t], false, false);
            }
        }
    }
    float inv[8];
#pragma unroll
    for (int r = 0; r < 8; ++r) inv[r] = 1.f / lst[r];
#pragma unroll
    for (int t = 0; t < 8; ++t)
#pragma unroll
        for (int r = 0; r < 8; ++r) {
            int srow = qrow0 + hi * 8 + r;
            attn[(size_t)(b * 2048 + srow) * 2048 + h * 128 + t * 16 + col] =
                f2bf(oacc[t][r] * inv[r]);
        }
}

// ---------------------------------------------------------------------------
extern "C" void kernel_launch(void* const* d_in, const int* in_sizes, int n_in,
                              void* d_out, int out_size, void* d_ws, size_t ws_size,
                              hipStream_t stream) {
    (void)in_sizes; (void)n_in; (void)out_size; (void)ws_size;
    const float* x   = (const float*)d_in[0];
    const float* pe  = (const float*)d_in[1];
    const float* q_w = (const float*)d_in[2];
    const float* k_w = (const float*)d_in[3];
    const float* v_w = (const float*)d_in[4];
    const float* o_w = (const float*)d_in[5];
    const float* qnw = (const float*)d_in[6];
    const float* knw = (const float*)d_in[7];

    char* w = (char*)d_ws;
    u16*   x_bf  = (u16*)w;   w += 16777216;   // x bf16          [4096,2048]
    u16*   wcat  = (u16*)w;   w += 16777216;   // [Wq;Wk;Wv] bf16 [4096,2048]
    u16*   ow_bf = (u16*)w;   w += 8388608;    // Wo bf16         [2048,2048]
    float* qkv   = (float*)w; w += 67108864;   // QKV fp32        [4096,4096]
    u16*   qb    = (u16*)w;   w += 16777216;   // Q bf16   [2,16,2048,128]
    u16*   kb_   = (u16*)w;   w += 8388608;    // K bf16   [2,8,2048,128]
    u16*   vtb   = (u16*)w;   w += 8388608;    // V^T bf16 [2,8,128,2048]
    u16*   attn  = (u16*)w;   w += 16777216;   // attn out bf16   [4096,2048]

    k_f2bf<<<8192, 256, 0, stream>>>(x, x_bf);
    k_wcat<<<8192, 256, 0, stream>>>(q_w, k_w, v_w, wcat);
    k_f2bf<<<4096, 256, 0, stream>>>(o_w, ow_bf);
    k_gemm<<<dim3(32, 32), 256, 0, stream>>>(x_bf, wcat, qkv, 4096, 4096, 2048);
    k_normrope<<<4096, 256, 0, stream>>>(qkv, pe, qnw, knw, qb, kb_, vtb);
    k_attn<<<dim3(16, 16, 2), 256, 0, stream>>>(qb, kb_, vtb, attn);
    k_gemm<<<dim3(16, 32), 256, 0, stream>>>(attn, ow_bf, (float*)d_out,
                                             4096, 2048, 2048);
}
